// ChiralMessage_33423435498372
// MI455X (gfx1250) — matrix operand order repaired and checked by reading, not compiled, not run on hardware
//
#include <hip/hip_runtime.h>
#include <stdint.h>

#define N_NODES 8192
#define KNBR 6
#define DIM 128
#define NTRIP 20
#define TOT_TRIPS (N_NODES * NTRIP) /* 163840 */
#define G_BLOCK 16                  /* triplets per workgroup */
#define ROWS 48                     /* 3 * G_BLOCK */
#define LSTRIDE 136                 /* padded LDS row stride in bf16 elems (272B: 4-bank skew) */
#define FST 132                     /* padded f32 LDS stride for the reduction buffer */
#define TWO_PI_F 6.2831853071795864769f

typedef __bf16 bf16;
typedef __attribute__((ext_vector_type(16))) __bf16 v16bf;
typedef __attribute__((ext_vector_type(8)))  float  v8f;
typedef __attribute__((ext_vector_type(8)))  unsigned int u32x8;
typedef __attribute__((ext_vector_type(4)))  unsigned int u32x4;
typedef __attribute__((ext_vector_type(8)))  int          i32x8;
typedef __attribute__((ext_vector_type(4)))  int          i32x4;

#if defined(__has_builtin)
#  if __has_builtin(__builtin_amdgcn_tensor_load_to_lds)
#    define HAVE_TDM 1
#  else
#    define HAVE_TDM 0
#  endif
#else
#  define HAVE_TDM 0
#endif

__constant__ int d_combos[20][3] = {
 {0,1,2},{0,1,3},{0,1,4},{0,1,5},{0,2,3},{0,2,4},{0,2,5},{0,3,4},{0,3,5},{0,4,5},
 {1,2,3},{1,2,4},{1,2,5},{1,3,4},{1,3,5},{1,4,5},{2,3,4},{2,3,5},{2,4,5},{3,4,5}};

// ---------- bf16 helpers (manual RNE, no libcalls) ----------
__device__ __forceinline__ float bf2f(bf16 x) {
    unsigned short s = __builtin_bit_cast(unsigned short, x);
    unsigned u = ((unsigned)s) << 16;
    return __builtin_bit_cast(float, u);
}
__device__ __forceinline__ bf16 f2bf(float f) {
    unsigned u = __builtin_bit_cast(unsigned, f);
    u += 0x7fffu + ((u >> 16) & 1u);
    unsigned short s = (unsigned short)(u >> 16);
    return __builtin_bit_cast(bf16, s);
}

// ---------- CDNA5 async global->LDS copies (ASYNCcnt-tracked) ----------
__device__ __forceinline__ unsigned lds_off(const void* p) {
    // generic LDS address: low 32 bits are the LDS byte offset
    return (unsigned)(unsigned long long)p;
}
__device__ __forceinline__ void async_b64(unsigned loff, const void* g) {
    asm volatile("global_load_async_to_lds_b64 %0, %1, off" :: "v"(loff), "v"(g) : "memory");
}
__device__ __forceinline__ void async_b128(unsigned loff, const void* g) {
    asm volatile("global_load_async_to_lds_b128 %0, %1, off" :: "v"(loff), "v"(g) : "memory");
}
#define WAIT_ASYNC_16() asm volatile("s_wait_asynccnt 16" ::: "memory")
#define WAIT_ASYNC_0()  asm volatile("s_wait_asynccnt 0"  ::: "memory")

// fallback: stage a 128x128 bf16 weight matrix into padded LDS via per-lane async copies
__device__ __forceinline__ void weight_async(bf16* dst_lds, const bf16* src, int tid) {
    #pragma unroll 1
    for (int c = tid; c < 2048; c += 128) {
        int row = c >> 4, c8 = c & 15;
        async_b128(lds_off(dst_lds + row * LSTRIDE + c8 * 8), src + c * 8);
    }
}

#if HAVE_TDM
// ---------- Tensor Data Mover: one-shot 128x128 bf16 tile load, HW-padded to LSTRIDE ----------
// D# group0: count=1 | lds_addr[63:32] | global_addr[120:64] | type=2 [127:126]
// D# group1: data_size=1(2B) | pad_enable, pad_interval=5 (64 DW = 256B row),
//            pad_amount=3 (4 DW = 16B)  -> reproduces the 272B bank-skewed row stride.
//            tensor_dim0=128, tensor_dim1=128, tile_dim0=128, tile_dim1=128, dim0_stride=128.
// This toolchain exposes the 6-arg builtin: (g0 x4, g1 x8, g2 x4, g3 x4, extra x8, cpol).
__device__ __forceinline__ void tdm_weight_load(bf16* dst_lds, const bf16* src) {
    unsigned long long va = (unsigned long long)src;
    u32x4 g0 = { 1u,
                 lds_off(dst_lds),
                 (unsigned)va,
                 (unsigned)((va >> 32) & 0x01ffffffull) | 0x80000000u };
    i32x8 g1 = { 0x07510000,   // data_size=1, pad_enable=1, pad_interval=5, pad_amount=3
                 0x00800000,   // tensor_dim0 = 128 (bits 79:48, low half)
                 0x00800000,   // tensor_dim0 hi=0 | tensor_dim1 = 128 (low half)
                 0x00800000,   // tensor_dim1 hi=0 | tile_dim0 = 128
                 0x00000080,   // tile_dim1 = 128, tile_dim2 = 0
                 0x00000080,   // tensor_dim0_stride = 128
                 0, 0 };       // stride hi, tensor_dim1_stride = 0 (2-D)
    i32x4 gz = { 0, 0, 0, 0 }; // groups 2/3: higher dims unused
    i32x8 gz8 = { 0, 0, 0, 0, 0, 0, 0, 0 };
    __builtin_amdgcn_tensor_load_to_lds(g0, g1, gz, gz, gz8, 0);
}
#endif

// ---------- WMMA fragment loaders (wave32 16x16x32 bf16 layouts) ----------
// A 16x32: lane m(0-15): K=k0+0..7 (V0-3), K=k0+16..23 (V4-7); lane m+16: +8
__device__ __forceinline__ v16bf frag_split(const bf16* p) {
    uint4 lo = *(const uint4*)p;        // 8 bf16 at p
    uint4 hi = *(const uint4*)(p + 16); // 8 bf16 at p+16 elems
    u32x8 u = {lo.x, lo.y, lo.z, lo.w, hi.x, hi.y, hi.z, hi.w};
    return __builtin_bit_cast(v16bf, u);
}
// 16 contiguous K values at p; unused (B fragments use frag_split, same K order as A)
__device__ __forceinline__ v16bf frag_cont(const bf16* p) {
    uint4 lo = *(const uint4*)p;
    uint4 hi = *(const uint4*)(p + 8);
    u32x8 u = {lo.x, lo.y, lo.z, lo.w, hi.x, hi.y, hi.z, hi.w};
    return __builtin_bit_cast(v16bf, u);
}
__device__ __forceinline__ v16bf load_afrag(const bf16* buf, int m0, int k0, int lane) {
    int m = m0 + (lane & 15);
    int h = lane >> 4;
    return frag_split(buf + m * LSTRIDE + k0 + 8 * h);
}
// fused msg = emb[PREV] - emb[NEXT] fragment (rows permute within a triplet)
__device__ __forceinline__ v16bf load_msg_afrag(const bf16* E, int m0, int k0, int lane) {
    int m = m0 + (lane & 15);
    int k = m % 3;
    int base = m - k;
    const int PREVk[3] = {2, 0, 1};
    const int NEXTk[3] = {1, 2, 0};
    int mP = base + PREVk[k], mN = base + NEXTk[k];
    int h = lane >> 4;
    v16bf A = frag_split(E + mP * LSTRIDE + k0 + 8 * h);
    v16bf B = frag_split(E + mN * LSTRIDE + k0 + 8 * h);
    v16bf r;
    #pragma unroll
    for (int i = 0; i < 16; ++i) r[i] = f2bf(bf2f(A[i]) - bf2f(B[i]));
    return r;
}
__device__ __forceinline__ v16bf load_bfrag(const bf16* w, int n0, int k0, int lane) {
    int n = n0 + (lane & 15);
    int kk = k0 + 8 * (lane >> 4);
    return frag_split(w + n * LSTRIDE + kk);
}

// ---------- one GEMM stage: [48x128] @ W^T(+bias)(+act) across the workgroup ----------
// Tile split: wave w owns N-columns {2w, 2w+1} x all 3 M-tiles, so the 4 A-fragments
// (incl. the VALU-heavy msg fragments) are hoisted and reused across the N loop.
// SRCM: 0 = read src buffer, 1 = msg (diff of sE rows)
// DSTM: 0 = store bf16 to dst; 1 = store (acc + sE) bf16 to dst; 2 = store f32 to fbuf
template <int SRCM, int DSTM, bool SILU>
__device__ __forceinline__ void gemm_stage(const bf16* w, const bf16* src, bf16* dst,
                                           const bf16* sE_, const float* __restrict__ bias,
                                           float* fbuf, int tid) {
    int lane = tid & 31, wave = tid >> 5;
    #pragma unroll 1
    for (int mi = 0; mi < 3; ++mi) {
        int m0 = mi * 16;
        v16bf a[4];
        #pragma unroll
        for (int kc = 0; kc < 4; ++kc)
            a[kc] = (SRCM == 0) ? load_afrag(src, m0, kc * 32, lane)
                                : load_msg_afrag(sE_, m0, kc * 32, lane);
        #pragma unroll
        for (int j = 0; j < 2; ++j) {
            int n0 = (wave * 2 + j) * 16;
            v8f c = {0.f, 0.f, 0.f, 0.f, 0.f, 0.f, 0.f, 0.f};
            #pragma unroll
            for (int kc = 0; kc < 4; ++kc) {
                v16bf b = load_bfrag(w, n0, kc * 32, lane);
                c = __builtin_amdgcn_wmma_f32_16x16x32_bf16(false, a[kc], false, b,
                                                            (short)0, c, false, false);
            }
            int n = n0 + (lane & 15);
            float bv = bias[n];
            int mbase = m0 + 8 * (lane >> 4); // C layout: VGPR r -> M = r (+8 for hi lanes)
            #pragma unroll
            for (int r = 0; r < 8; ++r) {
                float v = c[r] + bv;
                if (SILU) v = v / (1.f + __expf(-v));
                int m = mbase + r;
                if (DSTM == 0) {
                    dst[m * LSTRIDE + n] = f2bf(v);
                } else if (DSTM == 1) {
                    v += bf2f(sE_[m * LSTRIDE + n]);
                    dst[m * LSTRIDE + n] = f2bf(v);
                } else {
                    fbuf[m * FST + n] = v;
                }
            }
        }
    }
}

// ---------- fused MLP kernel ----------
__global__ void __launch_bounds__(128) mlp_kernel(const bf16* __restrict__ scb,
                                                  const int* __restrict__ oidx,
                                                  const bf16* __restrict__ wb,
                                                  const float* __restrict__ b1,
                                                  const float* __restrict__ b3,
                                                  const float* __restrict__ b4,
                                                  const float* __restrict__ b5,
                                                  const float* __restrict__ b6,
                                                  float* __restrict__ out) {
    __shared__ __attribute__((aligned(16))) bf16 sWA[DIM * LSTRIDE]; // 34.0 KB
    __shared__ __attribute__((aligned(16))) bf16 sWB[DIM * LSTRIDE]; // 34.0 KB (reused as f32 scratch)
    __shared__ __attribute__((aligned(16))) bf16 sX[ROWS * LSTRIDE]; // 12.75 KB
    __shared__ __attribute__((aligned(16))) bf16 sE[ROWS * LSTRIDE];
    __shared__ __attribute__((aligned(16))) bf16 sH[ROWS * LSTRIDE];
    float* fbuf = (float*)sWB; // 48*132*4 = 25.3 KB <= 34 KB; W5 is dead by stage 5

    int tid = threadIdx.x;
    int lane = tid & 31, wave = tid >> 5;
    int g0 = blockIdx.x * G_BLOCK;

    // async gather of 48 ordered node_scalar rows (8B per lane per row, ASYNCcnt)
    #pragma unroll 1
    for (int m = wave; m < ROWS; m += 4) {
        int gi = oidx[(g0 + m / 3) * 3 + (m % 3)];
        async_b64(lds_off(&sX[m * LSTRIDE + lane * 4]), scb + gi * DIM + lane * 4);
    }

#if HAVE_TDM
    if (wave == 0) {                                  // TDM: one descriptor per weight matrix
        tdm_weight_load(sWA, wb + 0 * DIM * DIM);     // W1
        tdm_weight_load(sWB, wb + 1 * DIM * DIM);     // W3 (prefetch)
    }
    WAIT_ASYNC_0();                                   // own gathers landed
    if (wave == 0) __builtin_amdgcn_s_wait_tensorcnt((short)1); // W1 landed (in-order)
#else
    weight_async(sWA, wb + 0 * DIM * DIM, tid);
    weight_async(sWB, wb + 1 * DIM * DIM, tid);
    WAIT_ASYNC_16();
#endif
    __syncthreads();

    gemm_stage<0, 0, false>(sWA, sX, sE, sE, b1, fbuf, tid);   // emb = X@W1^T + b1
    __syncthreads();

#if HAVE_TDM
    if (wave == 0) { tdm_weight_load(sWA, wb + 2 * DIM * DIM);             // W4 prefetch
                     __builtin_amdgcn_s_wait_tensorcnt((short)1); }        // W3 landed
#else
    weight_async(sWA, wb + 2 * DIM * DIM, tid);
    WAIT_ASYNC_16();
#endif
    __syncthreads();
    gemm_stage<1, 0, true>(sWB, sE, sH, sE, b3, fbuf, tid);    // h = silu(msg@W3^T + b3)
    __syncthreads();

#if HAVE_TDM
    if (wave == 0) { tdm_weight_load(sWB, wb + 3 * DIM * DIM);             // W5 prefetch
                     __builtin_amdgcn_s_wait_tensorcnt((short)1); }        // W4 landed
#else
    weight_async(sWB, wb + 3 * DIM * DIM, tid);
    WAIT_ASYNC_16();
#endif
    __syncthreads();
    gemm_stage<0, 1, false>(sWA, sH, sX, sE, b4, fbuf, tid);   // X = emb + (h@W4^T + b4)
    __syncthreads();

#if HAVE_TDM
    if (wave == 0) { tdm_weight_load(sWA, wb + 4 * DIM * DIM);             // W6 prefetch
                     __builtin_amdgcn_s_wait_tensorcnt((short)1); }        // W5 landed
#else
    weight_async(sWA, wb + 4 * DIM * DIM, tid);
    WAIT_ASYNC_16();
#endif
    __syncthreads();
    gemm_stage<0, 0, true>(sWB, sX, sH, sE, b5, fbuf, tid);    // h = silu(X@W5^T + b5)
    __syncthreads();

#if HAVE_TDM
    if (wave == 0) __builtin_amdgcn_s_wait_tensorcnt((short)0);            // W6 landed
#else
    WAIT_ASYNC_0();
#endif
    __syncthreads();
    gemm_stage<0, 2, false>(sWA, sH, sH, sE, b6, fbuf, tid);   // f32 tile results -> LDS
    __syncthreads();

    // per-block reduction: 48 rows -> <=2 node rows (a 16-triplet block spans <=2 nodes),
    // then <=2 global f32 atomics per thread instead of 48.
    {
        int c = tid; // 128 threads == 128 columns
        int nodeA = g0 / NTRIP;
        int nodeB = (g0 + G_BLOCK - 1) / NTRIP;
        float accA = 0.f, accB = 0.f;
        #pragma unroll 1
        for (int t2 = 0; t2 < G_BLOCK; ++t2) {
            float s = fbuf[(3 * t2 + 0) * FST + c]
                    + fbuf[(3 * t2 + 1) * FST + c]
                    + fbuf[(3 * t2 + 2) * FST + c];
            bool isA = ((g0 + t2) / NTRIP) == nodeA;
            accA += isA ? s : 0.f;
            accB += isA ? 0.f : s;
        }
        atomicAdd(out + nodeA * DIM + c, accA);
        if (nodeB != nodeA) atomicAdd(out + nodeB * DIM + c, accB);
    }
}

// ---------- geometric ordering kernel ----------
__global__ void order_kernel(const float* __restrict__ pos, const int* __restrict__ eidx,
                             int* __restrict__ oidx) {
    int g = blockIdx.x * blockDim.x + threadIdx.x;
    if (g >= TOT_TRIPS) return;
    int node = g / NTRIP, t = g - node * NTRIP;
    int s0 = d_combos[t][0], s1 = d_combos[t][1], s2 = d_combos[t][2];
    int i0 = eidx[(node * KNBR + s0) * 2 + 1];
    int i1 = eidx[(node * KNBR + s1) * 2 + 1];
    int i2 = eidx[(node * KNBR + s2) * 2 + 1];
    float p0x = pos[i0*3], p0y = pos[i0*3+1], p0z = pos[i0*3+2];
    float p1x = pos[i1*3], p1y = pos[i1*3+1], p1z = pos[i1*3+2];
    float p2x = pos[i2*3], p2y = pos[i2*3+1], p2z = pos[i2*3+2];
    float bx = pos[node*3], by = pos[node*3+1], bz = pos[node*3+2];
    float cx = (p0x+p1x+p2x)*(1.f/3.f), cy = (p0y+p1y+p2y)*(1.f/3.f), cz = (p0z+p1z+p2z)*(1.f/3.f);
    float v0x=p0x-cx, v0y=p0y-cy, v0z=p0z-cz;
    float v1x=p1x-cx, v1y=p1y-cy, v1z=p1z-cz;
    float v2x=p2x-cx, v2y=p2y-cy, v2z=p2z-cz;
    float e1x=v1x-v0x, e1y=v1y-v0y, e1z=v1z-v0z;
    float e2x=v2x-v0x, e2y=v2y-v0y, e2z=v2z-v0z;
    float nx = e1y*e2z - e1z*e2y, ny = e1z*e2x - e1x*e2z, nz = e1x*e2y - e1y*e2x;
    float bvx = bx-cx, bvy = by-cy, bvz = bz-cz;
    if (nx*bvx + ny*bvy + nz*bvz < 0.f) { nx=-nx; ny=-ny; nz=-nz; }
    float ninv = __frsqrt_rn(nx*nx + ny*ny + nz*nz);
    nx*=ninv; ny*=ninv; nz*=ninv;
    float xinv = __frsqrt_rn(e1x*e1x + e1y*e1y + e1z*e1z);
    float xax=e1x*xinv, xay=e1y*xinv, xaz=e1z*xinv;
    float yax = ny*xaz - nz*xay, yay = nz*xax - nx*xaz, yaz = nx*xay - ny*xax;
    float a0 = fmodf(atan2f(v0x*yax+v0y*yay+v0z*yaz, v0x*xax+v0y*xay+v0z*xaz)+TWO_PI_F, TWO_PI_F);
    float a1 = fmodf(atan2f(v1x*yax+v1y*yay+v1z*yaz, v1x*xax+v1y*xay+v1z*xaz)+TWO_PI_F, TWO_PI_F);
    float a2 = fmodf(atan2f(v2x*yax+v2y*yay+v2z*yaz, v2x*xax+v2y*xay+v2z*xaz)+TWO_PI_F, TWO_PI_F);
    int trip[3] = {i0, i1, i2};
    int o0 = 0, o1 = 1, o2 = 2;
    float t0 = a0, t1 = a1, t2 = a2;
    if (t1 < t0) { int oi=o0;o0=o1;o1=oi; float tf=t0;t0=t1;t1=tf; }
    if (t2 < t1) { int oi=o1;o1=o2;o2=oi; float tf=t1;t1=t2;t2=tf;
                   if (t1 < t0) { int oj=o0;o0=o1;o1=oj; float tg=t0;t0=t1;t1=tg; } }
    oidx[g*3+0] = trip[o0];
    oidx[g*3+1] = trip[o1];
    oidx[g*3+2] = trip[o2];
}

// ---------- conversion / init kernels ----------
__global__ void cvt_sc_kernel(const float* __restrict__ src, bf16* __restrict__ dst) {
    int i = blockIdx.x * 256 + threadIdx.x;
    if (i < N_NODES * DIM) dst[i] = f2bf(src[i]);
}
__global__ void cvt_w_kernel(const float* __restrict__ w1, const float* __restrict__ w3,
                             const float* __restrict__ w4, const float* __restrict__ w5,
                             const float* __restrict__ w6, bf16* __restrict__ dst) {
    int i = blockIdx.x * 256 + threadIdx.x;
    if (i >= 5 * DIM * DIM) return;
    int m = i >> 14, r = i & (DIM * DIM - 1);
    const float* s = (m == 0) ? w1 : (m == 1) ? w3 : (m == 2) ? w4 : (m == 3) ? w5 : w6;
    dst[i] = f2bf(s[r]);
}
__global__ void init_out_kernel(const float* __restrict__ nc, float* __restrict__ out) {
    int i = blockIdx.x * 256 + threadIdx.x;
    if (i < N_NODES * DIM) out[i] = nc[i];
}

extern "C" void kernel_launch(void* const* d_in, const int* in_sizes, int n_in,
                              void* d_out, int out_size, void* d_ws, size_t ws_size,
                              hipStream_t stream) {
    const float* node_scalar = (const float*)d_in[0];
    const float* node_chiral = (const float*)d_in[1];
    const float* pos         = (const float*)d_in[2];
    const int*   eidx        = (const int*)d_in[3];
    const float* W1 = (const float*)d_in[4];  const float* b1 = (const float*)d_in[5];
    const float* W3 = (const float*)d_in[6];  const float* b3 = (const float*)d_in[7];
    const float* W4 = (const float*)d_in[8];  const float* b4 = (const float*)d_in[9];
    const float* W5 = (const float*)d_in[10]; const float* b5 = (const float*)d_in[11];
    const float* W6 = (const float*)d_in[12]; const float* b6 = (const float*)d_in[13];
    float* out = (float*)d_out;

    char* ws = (char*)d_ws;
    int*  oidx = (int*)ws;                     // 163840*3*4 B ~= 1.97 MB
    bf16* scb  = (bf16*)(ws + (2u << 20));     // N*128 bf16 = 2 MB
    bf16* wbb  = (bf16*)(ws + (4u << 20));     // 5*128*128 bf16 = 160 KB

    init_out_kernel<<<(N_NODES * DIM) / 256, 256, 0, stream>>>(node_chiral, out);
    cvt_sc_kernel<<<(N_NODES * DIM) / 256, 256, 0, stream>>>(node_scalar, scb);
    cvt_w_kernel<<<(5 * DIM * DIM + 255) / 256, 256, 0, stream>>>(W1, W3, W4, W5, W6, wbb);
    order_kernel<<<(TOT_TRIPS + 255) / 256, 256, 0, stream>>>(pos, eidx, oidx);
    mlp_kernel<<<TOT_TRIPS / G_BLOCK, 128, 0, stream>>>(scb, oidx, wbb, b1, b3, b4, b5, b6, out);
}
